// Holo_19928648254209
// MI455X (gfx1250) — compile-verified, matrix operand order checked by reading, not run on hardware
//
#include <hip/hip_runtime.h>
#include <hip/hip_bf16.h>

typedef __attribute__((ext_vector_type(16))) __bf16 v16bf;
typedef __attribute__((ext_vector_type(8)))  float  v8f;

#define N_NODES 4096
#define F_IN    255
#define HID     256
#define DDIM    256
#define NB      8
#define T_TUP   16384
#define NG      4
#define LN_EPS  1e-5f

// Async global->LDS staging path (CDNA5 GLOBAL_LOAD_ASYNC_TO_LDS_B128,
// tracked by ASYNCcnt). Builtin signature (from hipcc diagnostic):
//   param0 = AS1 (global) pointer to int __attribute__((vector_size(16)))
//   param1 = LDS pointer, param2/3 = imm offset / cpol
#if defined(__AMDGCN__) && __has_builtin(__builtin_amdgcn_global_load_async_to_lds_b128)
#define USE_ASYNC_LDS 1
#else
#define USE_ASYNC_LDS 0
#endif

#if USE_ASYNC_LDS
typedef int v4i __attribute__((vector_size(16)));
typedef __attribute__((address_space(1))) v4i gas_v4i;
typedef __attribute__((address_space(3))) v4i las_v4i;
__device__ __forceinline__ void async_g2l_b128(const __bf16* g, __bf16* l) {
  __builtin_amdgcn_global_load_async_to_lds_b128((gas_v4i*)g, (las_v4i*)l, 0, 0);
}
__device__ __forceinline__ void wait_async0() {
#if __has_builtin(__builtin_amdgcn_s_wait_asynccnt)
  __builtin_amdgcn_s_wait_asynccnt(0);
#else
  asm volatile("s_wait_asynccnt 0x0" ::: "memory");
#endif
}
#endif

// ---------------------------------------------------------------------------
// deg[row] = sum_m adj[row][m]
// ---------------------------------------------------------------------------
__global__ __launch_bounds__(256) void deg_kernel(const float* __restrict__ adj,
                                                  float* __restrict__ deg) {
  __shared__ float red[256];
  const int row = blockIdx.x;
  float s = 0.f;
  for (int i = threadIdx.x; i < N_NODES; i += 256)
    s += adj[(size_t)row * N_NODES + i];
  red[threadIdx.x] = s;
  __syncthreads();
  for (int off = 128; off > 0; off >>= 1) {
    if (threadIdx.x < off) red[threadIdx.x] += red[threadIdx.x + off];
    __syncthreads();
  }
  if (threadIdx.x == 0) deg[row] = red[0];
}

// ---------------------------------------------------------------------------
// top-8 of deg (lowest index wins ties, matching lax.top_k)
// ---------------------------------------------------------------------------
__global__ __launch_bounds__(256) void topk8_kernel(const float* __restrict__ deg,
                                                    int* __restrict__ brk) {
  __shared__ float sv[256];
  __shared__ int   si[256];
  __shared__ int   chosen[NB];
  for (int it = 0; it < NB; ++it) {
    float best = -3.0e38f;
    int   bi   = N_NODES;
    for (int i = threadIdx.x; i < N_NODES; i += 256) {
      bool skip = false;
      for (int c = 0; c < it; ++c)
        if (chosen[c] == i) skip = true;
      float v = deg[i];
      if (!skip && (v > best || (v == best && i < bi))) { best = v; bi = i; }
    }
    sv[threadIdx.x] = best;
    si[threadIdx.x] = bi;
    __syncthreads();
    for (int off = 128; off > 0; off >>= 1) {
      if (threadIdx.x < off) {
        float ov = sv[threadIdx.x + off];
        int   oi = si[threadIdx.x + off];
        if (ov > sv[threadIdx.x] || (ov == sv[threadIdx.x] && oi < si[threadIdx.x])) {
          sv[threadIdx.x] = ov;
          si[threadIdx.x] = oi;
        }
      }
      __syncthreads();
    }
    if (threadIdx.x == 0) { chosen[it] = si[0]; brk[it] = si[0]; }
    __syncthreads();
  }
}

// ---------------------------------------------------------------------------
// elementwise f32 -> bf16 (grid-stride) — used for adj (stays row-major A)
// ---------------------------------------------------------------------------
__global__ __launch_bounds__(256) void cvt_bf16_kernel(const float* __restrict__ in,
                                                       __bf16* __restrict__ out,
                                                       long n) {
  for (long i = (long)blockIdx.x * 256 + threadIdx.x; i < n; i += (long)gridDim.x * 256)
    out[i] = (__bf16)in[i];
}

// ---------------------------------------------------------------------------
// tiled transpose + f32->bf16: in (R x C) row-major -> out (C x R) bf16
// ---------------------------------------------------------------------------
__global__ __launch_bounds__(256) void transpose_cvt_kernel(const float* __restrict__ in,
                                                            __bf16* __restrict__ out,
                                                            int R, int C) {
  __shared__ float tile[32][33];
  const int bx = blockIdx.x * 32;  // C offset
  const int by = blockIdx.y * 32;  // R offset
  const int tx = threadIdx.x & 31, ty = threadIdx.x >> 5;
#pragma unroll
  for (int i = 0; i < 32; i += 8)
    tile[ty + i][tx] = in[(size_t)(by + ty + i) * C + bx + tx];
  __syncthreads();
#pragma unroll
  for (int i = 0; i < 32; i += 8)
    out[(size_t)(bx + ty + i) * R + by + tx] = (__bf16)tile[tx][ty + i];
}

// ---------------------------------------------------------------------------
// X (4096x255 f32) -> Xt (256x4096 bf16), feature 255 zero-padded, transposed
// ---------------------------------------------------------------------------
__global__ __launch_bounds__(256) void pack_xt_kernel(const float* __restrict__ X,
                                                      __bf16* __restrict__ Xt) {
  __shared__ float tile[32][33];
  const int bx = blockIdx.x * 32;  // feature
  const int by = blockIdx.y * 32;  // node
  const int tx = threadIdx.x & 31, ty = threadIdx.x >> 5;
#pragma unroll
  for (int i = 0; i < 32; i += 8) {
    const int f = bx + tx, n = by + ty + i;
    tile[ty + i][tx] = (f < F_IN) ? X[(size_t)n * F_IN + f] : 0.f;
  }
  __syncthreads();
#pragma unroll
  for (int i = 0; i < 32; i += 8)
    Xt[(size_t)(bx + ty + i) * N_NODES + by + tx] = (__bf16)tile[tx][ty + i];
}

// ---------------------------------------------------------------------------
// Tiled bf16 GEMM with B pre-transposed: C = A(MxK) * Bt(NxK)^T [+bias][relu]
//   Block tile 128x128, K-step 32, 8 waves (4x2), wave tile 32x64.
//   Double-buffered LDS, one barrier per k-step. Tiles stream straight to LDS
//   via GLOBAL_LOAD_ASYNC_TO_LDS_B128 (fence: s_wait_asynccnt 0); fallback
//   path bounces tiles via VGPRs.
// EPI: 0 = store f32, 1 = store bf16, 2 = +bias,relu,bf16, 3 = +bias,f32
// ---------------------------------------------------------------------------
template <int EPI>
__global__ __launch_bounds__(256) void gemm_bf16_kernel(
    const __bf16* __restrict__ A, const __bf16* __restrict__ Bt,
    const float* __restrict__ bias, void* __restrict__ Cout,
    int M, int N, int K, size_t strideB, size_t strideC) {
  constexpr int LDT  = 40;        // padded LDS row stride (bf16); 16B-aligned rows
  constexpr int BUFO = 128 * LDT; // elems between double buffers
  __shared__ __bf16 sA[2][BUFO];
  __shared__ __bf16 sB[2][BUFO];

  const int t  = threadIdx.x;
  const int m0 = blockIdx.y * 128;
  const int n0 = blockIdx.x * 128;
  const __bf16* Bb = Bt + strideB * blockIdx.z;

  const int wave  = t >> 5;
  const int lane  = t & 31;
  const int wm    = (wave & 3) * 32;   // wave M offset in tile
  const int wn    = (wave >> 2) * 64;  // wave N offset in tile
  const int lrow  = lane & 15;
  const int khalf = (lane >> 4) << 4;  // 0 or 16

  // staging: each thread owns one contiguous 16-elem (32B) chunk per matrix
  const int srow = t >> 1, scol = (t & 1) << 4;  // 128 rows x 32 k
  const __bf16* gA = A  + (size_t)(m0 + srow) * K + scol;
  const __bf16* gB = Bb + (size_t)(n0 + srow) * K + scol;
  __bf16* lA = &sA[0][srow * LDT + scol];
  __bf16* lB = &sB[0][srow * LDT + scol];

  v8f acc[2][4] = {};
  const int ktiles = K >> 5;

#if USE_ASYNC_LDS
  auto stage_async = [&](int buf, int kt) {
    const __bf16* ga = gA + (size_t)kt * 32;
    const __bf16* gb = gB + (size_t)kt * 32;
    __bf16* la = lA + buf * BUFO;
    __bf16* lb = lB + buf * BUFO;
    async_g2l_b128(ga, la);
    async_g2l_b128(ga + 8, la + 8);
    async_g2l_b128(gb, lb);
    async_g2l_b128(gb + 8, lb + 8);
  };
  stage_async(0, 0);
  wait_async0();
  __syncthreads();
#else
  v16bf av = *(const v16bf*)gA;
  v16bf bv = *(const v16bf*)gB;
  *(v16bf*)lA = av;
  *(v16bf*)lB = bv;
  __syncthreads();
#endif

  for (int kt = 0; kt < ktiles; ++kt) {
    const int  cur  = kt & 1;
    const bool more = (kt + 1) < ktiles;
    if (more) {
#if USE_ASYNC_LDS
      stage_async(cur ^ 1, kt + 1);  // streams into LDS while WMMAs run
#else
      av = *(const v16bf*)(gA + (size_t)(kt + 1) * 32);
      bv = *(const v16bf*)(gB + (size_t)(kt + 1) * 32);
#endif
      if (kt + 2 < ktiles)  // L2 prefetch two tiles ahead on the big operand
        __builtin_prefetch(gA + (size_t)(kt + 2) * 32, 0, 1);
    }

    v16bf af[2], bfm[4];
    af[0] = *(const v16bf*)&sA[cur][(wm + lrow) * LDT + khalf];
    af[1] = *(const v16bf*)&sA[cur][(wm + 16 + lrow) * LDT + khalf];
#pragma unroll
    for (int j = 0; j < 4; ++j)
      bfm[j] = *(const v16bf*)&sB[cur][(wn + j * 16 + lrow) * LDT + khalf];

#pragma unroll
    for (int i = 0; i < 2; ++i)
#pragma unroll
      for (int j = 0; j < 4; ++j)
        acc[i][j] = __builtin_amdgcn_wmma_f32_16x16x32_bf16(
            false, af[i], false, bfm[j], (short)0, acc[i][j], false, false);

    if (more) {
#if USE_ASYNC_LDS
      wait_async0();
#else
      *(v16bf*)(lA + (cur ^ 1) * BUFO) = av;
      *(v16bf*)(lB + (cur ^ 1) * BUFO) = bv;
#endif
      __syncthreads();
    }
  }

  // epilogue: D layout — VGPR r: M = r (+8 for lanes>=16); N = lane%16
  const int mrow0 = m0 + wm + ((lane >> 4) << 3);
  const int ncol0 = n0 + wn + lrow;
#pragma unroll
  for (int i = 0; i < 2; ++i) {
#pragma unroll
    for (int j = 0; j < 4; ++j) {
#pragma unroll
      for (int r = 0; r < 8; ++r) {
        const int m = mrow0 + i * 16 + r;
        const int n = ncol0 + j * 16;
        float v = acc[i][j][r];
        const size_t off = strideC * blockIdx.z + (size_t)m * N + n;
        if constexpr (EPI == 0) {
          ((float*)Cout)[off] = v;
        } else if constexpr (EPI == 1) {
          ((__bf16*)Cout)[off] = (__bf16)v;
        } else if constexpr (EPI == 2) {
          v += bias[n];
          ((__bf16*)Cout)[off] = (__bf16)fmaxf(v, 0.f);
        } else {
          v += bias[n];
          ((float*)Cout)[off] = v;
        }
      }
    }
  }
}

// ---------------------------------------------------------------------------
// Ht[b][j][n] = relu( Z[n][j] + adj[n][brk[b]] * W1[255][j] + b1[j] ) -> bf16
// feature-major output (ready as Bt for the batched GEMM)
// ---------------------------------------------------------------------------
__global__ __launch_bounds__(256) void build_ht_kernel(
    const float* __restrict__ Z, const float* __restrict__ adj,
    const int* __restrict__ brk, const float* __restrict__ W1,
    const float* __restrict__ b1, __bf16* __restrict__ Ht) {
  __shared__ float tile[32][33];
  const int b  = blockIdx.z;
  const int bj = blockIdx.x * 32;  // feature
  const int bn = blockIdx.y * 32;  // node
  const int tx = threadIdx.x & 31, ty = threadIdx.x >> 5;
  const int brkb = brk[b];
#pragma unroll
  for (int i = 0; i < 32; i += 8) {
    const int n = bn + ty + i, j = bj + tx;
    const float col = adj[(size_t)n * N_NODES + brkb];
    const float v = Z[(size_t)n * HID + j] + col * W1[(size_t)F_IN * HID + j] + b1[j];
    tile[ty + i][tx] = fmaxf(v, 0.f);
  }
  __syncthreads();
#pragma unroll
  for (int i = 0; i < 32; i += 8)
    Ht[((size_t)b * HID + bj + ty + i) * N_NODES + bn + tx] = (__bf16)tile[tx][ty + i];
}

// ---------------------------------------------------------------------------
// in-place LayerNorm over last dim (256)
// ---------------------------------------------------------------------------
__global__ __launch_bounds__(256) void ln_kernel(float* __restrict__ OUT,
                                                 const float* __restrict__ gamma,
                                                 const float* __restrict__ beta) {
  __shared__ float s1[256], s2[256];
  const size_t row = blockIdx.x;
  const int j = threadIdx.x;
  const float v = OUT[row * DDIM + j];
  s1[j] = v;
  s2[j] = v * v;
  __syncthreads();
  for (int off = 128; off > 0; off >>= 1) {
    if (j < off) { s1[j] += s1[j + off]; s2[j] += s2[j + off]; }
    __syncthreads();
  }
  const float mu  = s1[0] * (1.f / DDIM);
  const float var = s2[0] * (1.f / DDIM) - mu * mu;
  OUT[row * DDIM + j] = (v - mu) * rsqrtf(var + LN_EPS) * gamma[j] + beta[j];
}

// ---------------------------------------------------------------------------
// set_repr[b,t,:] = holo[b,i0,:]*holo[b,i1,:]; segment-mean over b by group
// ---------------------------------------------------------------------------
__global__ __launch_bounds__(256) void final_kernel(
    const float* __restrict__ holo, const int* __restrict__ tup,
    const int* __restrict__ gidx, float* __restrict__ out) {
  const int t = blockIdx.x, j = threadIdx.x;
  const int i0 = tup[t];
  const int i1 = tup[T_TUP + t];
  float accg[NG] = {0.f, 0.f, 0.f, 0.f};
  float cnt[NG]  = {0.f, 0.f, 0.f, 0.f};
#pragma unroll
  for (int b = 0; b < NB; ++b) {
    const int g = gidx[b];
    const float s = holo[((size_t)b * N_NODES + i0) * DDIM + j] *
                    holo[((size_t)b * N_NODES + i1) * DDIM + j];
#pragma unroll
    for (int gg = 0; gg < NG; ++gg)
      if (gg == g) { accg[gg] += s; cnt[gg] += 1.f; }
  }
#pragma unroll
  for (int g = 0; g < NG; ++g)
    out[(size_t)t * (NG * DDIM) + g * DDIM + j] = accg[g] / fmaxf(cnt[g], 1.f);
}

// ---------------------------------------------------------------------------
extern "C" void kernel_launch(void* const* d_in, const int* in_sizes, int n_in,
                              void* d_out, int out_size, void* d_ws, size_t ws_size,
                              hipStream_t stream) {
  const float* X   = (const float*)d_in[0];
  const float* adj = (const float*)d_in[1];
  const float* W1  = (const float*)d_in[2];
  const float* b1  = (const float*)d_in[3];
  const float* W2  = (const float*)d_in[4];
  const float* b2  = (const float*)d_in[5];
  const float* lng = (const float*)d_in[6];
  const float* lnb = (const float*)d_in[7];
  const int* tup   = (const int*)d_in[8];
  const int* gidx  = (const int*)d_in[9];
  float* out = (float*)d_out;

  char* ws = (char*)d_ws;
  auto alloc = [&](size_t bytes) -> char* {
    char* p = ws;
    ws += (bytes + 255) & ~(size_t)255;
    return p;
  };
  const size_t NN = (size_t)N_NODES * N_NODES;          // 16.7M
  const size_t NH = (size_t)N_NODES * HID;              // 1M
  __bf16* adjB = (__bf16*)alloc(NN * 2);                // 32 MB, row-major (A)
  __bf16* Xt   = (__bf16*)alloc(NH * 2);                // 2 MB, 256x4096 (Bt)
  __bf16* W1t  = (__bf16*)alloc((size_t)HID * HID * 2); // 256x256 transposed
  __bf16* W2t  = (__bf16*)alloc((size_t)HID * DDIM * 2);
  __bf16* AXb  = (__bf16*)alloc(NH * 2);                // 2 MB, 4096x256 (A)
  float*  Z    = (float*) alloc(NH * 4);                // 4 MB
  __bf16* Ht   = (__bf16*)alloc((size_t)NB * NH * 2);   // 16 MB, [b][256][4096]
  __bf16* P    = (__bf16*)alloc((size_t)NB * NH * 2);   // 16 MB, [b][4096][256]
  float*  OUTb = (float*) alloc((size_t)NB * NH * 4);   // 32 MB
  float*  deg  = (float*) alloc((size_t)N_NODES * 4);
  int*    brk  = (int*)   alloc(64);

  // 1) degree + top-8
  deg_kernel<<<N_NODES, 256, 0, stream>>>(adj, deg);
  topk8_kernel<<<1, 256, 0, stream>>>(deg, brk);

  // 2) precision conversions / layout prep
  cvt_bf16_kernel<<<4096, 256, 0, stream>>>(adj, adjB, (long)NN);
  pack_xt_kernel<<<dim3(HID / 32, N_NODES / 32), 256, 0, stream>>>(X, Xt);
  transpose_cvt_kernel<<<dim3(8, 8), 256, 0, stream>>>(W1, W1t, HID, HID);
  transpose_cvt_kernel<<<dim3(8, 8), 256, 0, stream>>>(W2, W2t, HID, DDIM);

  // 3) shared layer-1: AX = adj @ Xp (bf16 out), Z = AX @ W1 (f32 out)
  gemm_bf16_kernel<1><<<dim3(2, 32, 1), 256, 0, stream>>>(
      adjB, Xt, nullptr, AXb, N_NODES, HID, N_NODES, 0, 0);
  gemm_bf16_kernel<0><<<dim3(2, 32, 1), 256, 0, stream>>>(
      AXb, W1t, nullptr, Z, N_NODES, HID, HID, 0, 0);

  // 4) per-b rank-1 correction + bias + relu -> Ht (feature-major bf16)
  build_ht_kernel<<<dim3(HID / 32, N_NODES / 32, NB), 256, 0, stream>>>(
      Z, adj, brk, W1, b1, Ht);

  // 5) dominant batched GEMM: P[b] = adj @ H[b]  (8 x 4096x4096x256)
  gemm_bf16_kernel<1><<<dim3(2, 32, NB), 256, 0, stream>>>(
      adjB, Ht, nullptr, P, N_NODES, HID, N_NODES, NH, NH);

  // 6) OUT = P @ W2 + b2 (flattened M = 8*4096)
  gemm_bf16_kernel<3><<<dim3(2, 256, 1), 256, 0, stream>>>(
      P, W2t, b2, OUTb, NB * N_NODES, DDIM, HID, 0, 0);

  // 7) LayerNorm in place
  ln_kernel<<<NB * N_NODES, 256, 0, stream>>>(OUTb, lng, lnb);

  // 8) gather pairs, product, segment mean, interleave output
  final_kernel<<<T_TUP, 256, 0, stream>>>(OUTb, tup, gidx, out);
}
